// PeakPredictorV2_85023172592366
// MI455X (gfx1250) — matrix operand order repaired and checked by reading, not compiled, not run on hardware
//
#include <hip/hip_runtime.h>

// ---------------------------------------------------------------------------
// PeakPredictorV2 fused implementation for gfx1250 (MI455X, wave32, WMMA).
// Tiny dense layers run as V_WMMA_F32_16X16X32_F16 tiles (2 row-tiles of 16
// over the 32 peaks), f16 in / f32 acc, BN folded once per launch.
// Weights are pre-swizzled into the ISA B-operand lane layout (f16) so each
// wave fetches a whole B operand with one 32B vector load per lane.
// Per-wave LDS staging relies on wave-internal DS ordering (in-order per ISA),
// guarded only by compiler fences (zero-cost wave barriers).
// ---------------------------------------------------------------------------

#define HH 23
#define WW 23
#define HW 529
#define PP 32
#define BT 4096
#define THR 0.05f
#define NEGL -1e9f
#define NEGINF -3.0e38f

typedef __attribute__((ext_vector_type(16))) _Float16 v16h;
typedef __attribute__((ext_vector_type(8)))  float    v8f;

// ----- d_in flatten order (jax pytree: dict keys sorted) --------------------
enum {
  IN_TS=0, IN_ANNO=1,
  D_B1=2, D_B2=3, D_B3=4,
  D_BN1_B=5, D_BN1_G=6, D_BN1_M=7, D_BN1_V=8,
  D_BN2_B=9, D_BN2_G=10, D_BN2_M=11, D_BN2_V=12,
  D_W1=13, D_W2=14, D_W3=15,
  E_B1=16, E_B2=17, E_BF=18,
  E_BN1_B=19, E_BN1_G=20, E_BN1_M=21, E_BN1_V=22,
  E_BN2_B=23, E_BN2_G=24, E_BN2_M=25, E_BN2_V=26,
  E_BW=27, E_W1=28, E_W2=29, E_WF=30, E_WW=31,
  NPAR=32
};
struct Params { const float* p[NPAR]; };

// ----- workspace layout (float units) ---------------------------------------
#define WS_C_OLD   0                      // int[B][P][2] (reinterpreted)
#define WS_S_OLD   (BT*PP*2)             // float[B][P]
#define WS_M_OLD   (WS_S_OLD + BT*PP)    // float[B][P] (0/1)
#define WS_EMB_OLD (WS_M_OLD + BT*PP)    // float[B][8]
#define WS_WTS     (WS_EMB_OLD + BT*8)   // folded scalar weights region
// folded scalar weights offsets (relative to WS_WTS)
#define OFF_EW1 0        // [8][4]
#define OFF_EB1 32       // [8]
#define OFF_EW2 40       // [8][8]
#define OFF_EB2 104      // [8]
#define OFF_EWF 112      // [9][8]  rows 0..7 = wf, row 8 = ww
#define OFF_EBF 184      // [9]
#define OFF_DW1 193      // [32][17]
#define OFF_DB1 737      // [32]
#define OFF_DW2 769      // [64][32]
#define OFF_DB2 2817     // [64]
#define OFF_DW3 2881     // [64]
#define OFF_DB3 2945     // [1]
// pre-swizzled f16 B-operand blocks: 11 blocks x 512 halves (32B aligned)
#define WS_WOPS (WS_WTS + 2952)          // float offset (multiple of 8)
enum { OPB_EW1=0, OPB_EW2=1, OPB_EWF=2, OPB_DW1A=3, OPB_DW1B=4,
       OPB_DW2_0=5, OPB_DW2_1=6, OPB_DW2_2=7, OPB_DW2_3=8,
       OPB_DW3A=9, OPB_DW3B=10, NOPB=11 };
// total ws: (WS_WOPS + NOPB*256) floats ~= 2.25 MB

// ----- d_out layout: [match_pred | c_cur | s_cur | m_cur], 5*B*P floats -----
#define MP_OFF 0
#define C_OFF  (BT*PP)
#define S_OFF  (BT*PP*3)
#define M_OFF  (BT*PP*4)

// ===========================================================================
// Helpers
// ===========================================================================

// Intra-wave LDS ordering: DS ops from one wave complete in order (ISA ch11);
// we only need to stop the compiler from reordering across the handoff.
__device__ inline void wsync() { __builtin_amdgcn_wave_barrier(); }

__device__ inline v8f zero8() {
  v8f z = {0.f,0.f,0.f,0.f,0.f,0.f,0.f,0.f};
  return z;
}

__device__ inline v8f wmma16(v16h a, v16h b, v8f c) {
  // v_wmma_f32_16x16x32_f16: (neg_a, A, neg_b, B, c_mod, C, reuseA, reuseB)
  return __builtin_amdgcn_wmma_f32_16x16x32_f16(false, a, false, b, (short)0, c,
                                                false, false);
}

// Load a pre-swizzled 32x16 f16 B operand: one 32B vector load per lane.
__device__ inline v16h load_bop(const _Float16* hw, int blk, int lane) {
  return *(const v16h*)(hw + blk * 512 + lane * 16);
}

// A-matrix 16x32 f16 tile from row-major f32 LDS buffer S[row*ld + k].
// lanes 0-15: M=lane, halves 0..7 -> K=0..7, halves 8..15 -> K=16..23
// lanes16-31: M=lane-16, halves 0..7 -> K=8..15, halves 8..15 -> K=24..31
// kvalid should be chosen so the guard is lane-uniform (16 or 32 here).
__device__ inline v16h make_a(const float* S, int ld, int row0, int kvalid, int lane) {
  int r = row0 + (lane & 15);
  int kbase = (lane & 16) ? 8 : 0;
  v16h a;
#pragma unroll
  for (int h = 0; h < 16; ++h) {
    int k = kbase + h + ((h & 8) ? 8 : 0);
    float v = (k < kvalid) ? S[r * ld + k] : 0.0f;
    a[h] = (_Float16)v;
  }
  return a;
}

// Layer-1 A operand (K=4) built from per-lane registers (row == peak == lane):
// tile rows row0..row0+15; only lanes 0-15 carry K=0..3, everything else 0.
__device__ inline v16h make_a_reg4(float f0, float f1, float f2, float f3,
                                   int row0, int lane) {
  int src = row0 + (lane & 15);
  float g0 = __shfl(f0, src, 32), g1 = __shfl(f1, src, 32);
  float g2 = __shfl(f2, src, 32), g3 = __shfl(f3, src, 32);
  bool lo = (lane < 16);
  v16h a;
#pragma unroll
  for (int h = 0; h < 16; ++h) a[h] = (_Float16)0.0f;
  a[0] = lo ? (_Float16)g0 : (_Float16)0.0f;
  a[1] = lo ? (_Float16)g1 : (_Float16)0.0f;
  a[2] = lo ? (_Float16)g2 : (_Float16)0.0f;
  a[3] = lo ? (_Float16)g3 : (_Float16)0.0f;
  return a;
}

// C/D 16x16 f32 -> row-major f32 LDS with pre-hoisted per-lane bias (+relu).
// lane holds column N=lane&15; VGPR r -> row (row0 + r + 8*(lane>=16)).
// Writes all 16 columns (zeros beyond nvalid) so later A reads are uniform.
__device__ inline void store_c(float* S, int ld, int row0, v8f acc,
                               float bv, bool relu_on, int lane) {
  int n = lane & 15;
  int rbase = row0 + ((lane & 16) ? 8 : 0);
#pragma unroll
  for (int r = 0; r < 8; ++r) {
    float v = acc[r] + bv;
    if (relu_on) v = fmaxf(v, 0.0f);
    S[(rbase + r) * ld + n] = v;
  }
}

// ===========================================================================
// Kernel 0: fold BatchNorm into dense weights/biases, then pre-swizzle the
// folded weights into f16 B-operand lane layout (11 blocks of 32 lanes x 16).
// ===========================================================================
__global__ void k_prep(Params pr, float* ws) {
  float* wt = ws + WS_WTS;
  _Float16* hw = (_Float16*)(ws + WS_WOPS);
  int t = threadIdx.x;
  // ---- pass 1: BN fold into scalar arrays ----
  for (int n = t; n < 8; n += blockDim.x) {
    float a1 = pr.p[E_BN1_G][n] * rsqrtf(pr.p[E_BN1_V][n] + 1e-5f);
    wt[OFF_EB1 + n] = (pr.p[E_B1][n] - pr.p[E_BN1_M][n]) * a1 + pr.p[E_BN1_B][n];
    for (int k = 0; k < 4; ++k) wt[OFF_EW1 + n*4 + k] = pr.p[E_W1][n*4 + k] * a1;
    float a2 = pr.p[E_BN2_G][n] * rsqrtf(pr.p[E_BN2_V][n] + 1e-5f);
    wt[OFF_EB2 + n] = (pr.p[E_B2][n] - pr.p[E_BN2_M][n]) * a2 + pr.p[E_BN2_B][n];
    for (int k = 0; k < 8; ++k) wt[OFF_EW2 + n*8 + k] = pr.p[E_W2][n*8 + k] * a2;
    wt[OFF_EBF + n] = pr.p[E_BF][n];
    for (int k = 0; k < 8; ++k) wt[OFF_EWF + n*8 + k] = pr.p[E_WF][n*8 + k];
  }
  if (t == 0) {
    wt[OFF_EBF + 8] = pr.p[E_BW][0];
    for (int k = 0; k < 8; ++k) wt[OFF_EWF + 8*8 + k] = pr.p[E_WW][k];
    wt[OFF_DB3] = pr.p[D_B3][0];
  }
  for (int n = t; n < 32; n += blockDim.x) {
    float a = pr.p[D_BN1_G][n] * rsqrtf(pr.p[D_BN1_V][n] + 1e-5f);
    wt[OFF_DB1 + n] = (pr.p[D_B1][n] - pr.p[D_BN1_M][n]) * a + pr.p[D_BN1_B][n];
    for (int k = 0; k < 17; ++k) wt[OFF_DW1 + n*17 + k] = pr.p[D_W1][n*17 + k] * a;
  }
  for (int n = t; n < 64; n += blockDim.x) {
    float a = pr.p[D_BN2_G][n] * rsqrtf(pr.p[D_BN2_V][n] + 1e-5f);
    wt[OFF_DB2 + n] = (pr.p[D_B2][n] - pr.p[D_BN2_M][n]) * a + pr.p[D_BN2_B][n];
    for (int k = 0; k < 32; ++k) wt[OFF_DW2 + n*32 + k] = pr.p[D_W2][n*32 + k] * a;
    wt[OFF_DW3 + n] = pr.p[D_W3][n];
  }
  __syncthreads();
  // ---- pass 2: swizzle into B-operand lane layout ----
  // B layout: N = lane&15; lanes 0-15 hold K=0..7,16..23; lanes 16-31 hold K=8..15,24..31.
#pragma unroll
  for (int blk = 0; blk < NOPB; ++blk) {
    int src, ldk, nv, kv;
    switch (blk) {
      case OPB_EW1:   src = OFF_EW1;             ldk = 4;  nv = 8;  kv = 4;  break;
      case OPB_EW2:   src = OFF_EW2;             ldk = 8;  nv = 8;  kv = 8;  break;
      case OPB_EWF:   src = OFF_EWF;             ldk = 8;  nv = 9;  kv = 8;  break;
      case OPB_DW1A:  src = OFF_DW1;             ldk = 17; nv = 16; kv = 17; break;
      case OPB_DW1B:  src = OFF_DW1 + 16*17;     ldk = 17; nv = 16; kv = 17; break;
      case OPB_DW2_0: src = OFF_DW2;             ldk = 32; nv = 16; kv = 32; break;
      case OPB_DW2_1: src = OFF_DW2 + 16*32;     ldk = 32; nv = 16; kv = 32; break;
      case OPB_DW2_2: src = OFF_DW2 + 32*32;     ldk = 32; nv = 16; kv = 32; break;
      case OPB_DW2_3: src = OFF_DW2 + 48*32;     ldk = 32; nv = 16; kv = 32; break;
      case OPB_DW3A:  src = OFF_DW3;             ldk = 64; nv = 1;  kv = 32; break;
      default:        src = OFF_DW3 + 32;        ldk = 64; nv = 1;  kv = 32; break;
    }
    for (int rem = t; rem < 512; rem += blockDim.x) {
      int lane = rem >> 4, h = rem & 15;
      int n = lane & 15;
      int k = ((lane & 16) ? 8 : 0) + h + ((h & 8) ? 8 : 0);
      float v = (n < nv && k < kv) ? wt[src + n*ldk + k] : 0.0f;
      hw[blk*512 + rem] = (_Float16)v;
    }
  }
}

// ===========================================================================
// Kernel 1: 5x5-NMS peak finding + top-32 (desc) per map.
// grid (B, 2): y=0 -> "old" peaks to ws;  y=1 -> "cur" peaks directly to d_out.
// ===========================================================================
__global__ __launch_bounds__(256) void k_peaks(const float* tscores, float* out, float* ws) {
  __shared__ float sm[HW];
  __shared__ float mval[HW];
  __shared__ float red[256];
  __shared__ int   redi[256];
  int b = blockIdx.x, which = blockIdx.y, tid = threadIdx.x;
  const float* sc = tscores + ((size_t)which * BT + b) * HW;
  int* ci = (int*)ws;

  for (int i = tid; i < HW; i += 256) sm[i] = sc[i];
  __syncthreads();
  for (int i = tid; i < HW; i += 256) {
    int r = i / WW, c = i % WW;
    float mx = NEGINF;
    for (int dr = -2; dr <= 2; ++dr) {
      int rr = r + dr; if (rr < 0 || rr >= HH) continue;
      for (int dc = -2; dc <= 2; ++dc) {
        int cc = c + dc; if (cc < 0 || cc >= WW) continue;
        mx = fmaxf(mx, sm[rr * WW + cc]);
      }
    }
    float s = sm[i];
    mval[i] = (s == mx && s > THR) ? s : NEGINF;
  }
  __syncthreads();

  for (int p = 0; p < PP; ++p) {
    float bv = NEGINF; int bi = HW;
    for (int i = tid; i < HW; i += 256) {
      float v = mval[i];
      if (v > bv || (v == bv && i < bi)) { bv = v; bi = i; }
    }
    red[tid] = bv; redi[tid] = bi;
    __syncthreads();
    for (int s2 = 128; s2 > 0; s2 >>= 1) {
      if (tid < s2) {
        if (red[tid+s2] > red[tid] ||
            (red[tid+s2] == red[tid] && redi[tid+s2] < redi[tid])) {
          red[tid] = red[tid+s2]; redi[tid] = redi[tid+s2];
        }
      }
      __syncthreads();
    }
    if (tid == 0) {
      float val = red[0]; int idx = redi[0];
      if (idx >= HW) idx = 0;
      bool valid = val > THR;
      float vv = valid ? val : 0.0f;
      int r = idx / WW, c = idx % WW;
      if (which == 0) {
        ci[(b*PP + p)*2 + 0] = r;
        ci[(b*PP + p)*2 + 1] = c;
        ws[WS_S_OLD + b*PP + p] = vv;
        ws[WS_M_OLD + b*PP + p] = valid ? 1.0f : 0.0f;
      } else {
        out[C_OFF + (b*PP + p)*2 + 0] = (float)r;
        out[C_OFF + (b*PP + p)*2 + 1] = (float)c;
        out[S_OFF + b*PP + p] = vv;
        out[M_OFF + b*PP + p] = valid ? 1.0f : 0.0f;
      }
      mval[idx] = NEGINF;
    }
    __syncthreads();
  }
}

// ===========================================================================
// Kernel 2: old branch. One wave32 per batch element.
// ===========================================================================
__global__ __launch_bounds__(256) void k_old(const float* anno, float* ws) {
  __shared__ float stage[8][PP][16];
  int wv = threadIdx.x >> 5, lane = threadIdx.x & 31;
  int b = blockIdx.x * 8 + wv;
  const float* wt = ws + WS_WTS;
  const _Float16* hw = (const _Float16*)(ws + WS_WOPS);
  const int* ci = (const int*)ws;
  int n = lane & 15;

  float eb1 = (n < 8) ? wt[OFF_EB1 + n] : 0.0f;
  float eb2 = (n < 8) ? wt[OFF_EB2 + n] : 0.0f;
  float ebf = (n < 9) ? wt[OFF_EBF + n] : 0.0f;

  // argmax of annotation map (first occurrence)
  const float* y = anno + (size_t)b * HW;
  float bv = NEGINF; int bi = 0;
  for (int i = lane; i < HW; i += 32) {
    float v = y[i];
    if (v > bv) { bv = v; bi = i; }
  }
  for (int off = 16; off; off >>= 1) {
    float ov = __shfl_xor(bv, off, 32); int oi = __shfl_xor(bi, off, 32);
    if (ov > bv || (ov == bv && oi < bi)) { bv = ov; bi = oi; }
  }
  int gr = bi / WW, gc = bi % WW;

  // lane = old peak j
  int cx = ci[(b*PP + lane)*2 + 0], cy = ci[(b*PP + lane)*2 + 1];
  float sj = ws[WS_S_OLD + b*PP + lane];
  float mj = ws[WS_M_OLD + b*PP + lane];
  int dx = gr - cx, dy = gc - cy;
  int d2 = (mj > 0.5f) ? (dx*dx + dy*dy) : (1 << 30);
  int bd = d2, gth = lane;
  for (int off = 16; off; off >>= 1) {
    int od = __shfl_xor(bd, off, 32); int og = __shfl_xor(gth, off, 32);
    if (od < bd || (od == bd && og < gth)) { bd = od; gth = og; }
  }
  int ccx = __shfl(cx, gth, 32), ccy = __shfl(cy, gth, 32);
  float cs = __shfl(sj, gth, 32);

  // vss_old row j = [ccx-cx, ccy-cy, center_score, s_j]  (kept in registers)
  float f0 = (float)(ccx - cx), f1 = (float)(ccy - cy), f2 = cs, f3 = sj;
  float* S = &stage[wv][0][0];

  // L1: 4 -> 8 (A from registers)
  v16h bw = load_bop(hw, OPB_EW1, lane);
  v16h a0 = make_a_reg4(f0, f1, f2, f3, 0, lane);
  v16h a1 = make_a_reg4(f0, f1, f2, f3, 16, lane);
  v8f c0 = wmma16(a0, bw, zero8()), c1 = wmma16(a1, bw, zero8());
  store_c(S, 16, 0, c0, eb1, true, lane);
  store_c(S, 16, 16, c1, eb1, true, lane);
  wsync();
  // L2: 8 -> 8 (cols 8..15 of S are zero -> kvalid=16 is exact & uniform)
  bw = load_bop(hw, OPB_EW2, lane);
  a0 = make_a(S, 16, 0, 16, lane); a1 = make_a(S, 16, 16, 16, lane);
  c0 = wmma16(a0, bw, zero8()); c1 = wmma16(a1, bw, zero8());
  wsync();
  store_c(S, 16, 0, c0, eb2, true, lane);
  store_c(S, 16, 16, c1, eb2, true, lane);
  wsync();
  // L3: 8 -> 9 (cols 0..7 feat, col 8 weight logit), no relu
  bw = load_bop(hw, OPB_EWF, lane);
  a0 = make_a(S, 16, 0, 16, lane); a1 = make_a(S, 16, 16, 16, lane);
  c0 = wmma16(a0, bw, zero8()); c1 = wmma16(a1, bw, zero8());
  wsync();
  store_c(S, 16, 0, c0, ebf, false, lane);
  store_c(S, 16, 16, c1, ebf, false, lane);
  wsync();

  // masked softmax over j, embed_old = sum_j w_j * feat_j
  float wl = S[lane*16 + 8];
  wl = (mj > 0.5f) ? wl : NEGL;
  float mx = wl;
  for (int off = 16; off; off >>= 1) mx = fmaxf(mx, __shfl_xor(mx, off, 32));
  float e = __expf(wl - mx), se = e;
  for (int off = 16; off; off >>= 1) se += __shfl_xor(se, off, 32);
  float wj = e / se;
  float acc8[8];
#pragma unroll
  for (int k = 0; k < 8; ++k) acc8[k] = wj * S[lane*16 + k];
  for (int off = 16; off; off >>= 1) {
#pragma unroll
    for (int k = 0; k < 8; ++k) acc8[k] += __shfl_xor(acc8[k], off, 32);
  }
  if (lane == 0) {
#pragma unroll
    for (int k = 0; k < 8; ++k) ws[WS_EMB_OLD + b*8 + k] = acc8[k];
  }
}

// ===========================================================================
// Kernel 3: cur branch + discriminator. One wave32 per batch element (4/block).
// ===========================================================================
__global__ __launch_bounds__(128) void k_cur(float* out, float* ws) {
  __shared__ float U[4][PP][64];   // staging (embedding stage / disc L2 out)
  __shared__ float F[4][PP][32];   // feats17 (zero-padded) / disc L1 out
  int wv = threadIdx.x >> 5, lane = threadIdx.x & 31;
  int b = blockIdx.x * 4 + wv;
  const float* wt = ws + WS_WTS;
  const _Float16* hw = (const _Float16*)(ws + WS_WOPS);
  float* Ub = &U[wv][0][0];
  float* Fb = &F[wv][0][0];
  int n = lane & 15;

  // hoisted per-lane folded biases (loop-invariant)
  float eb1 = (n < 8) ? wt[OFF_EB1 + n] : 0.0f;
  float eb2 = (n < 8) ? wt[OFF_EB2 + n] : 0.0f;
  float ebf = (n < 9) ? wt[OFF_EBF + n] : 0.0f;
  float db1a = wt[OFF_DB1 + n];
  float db1b = wt[OFF_DB1 + 16 + n];
  float db2v[4];
#pragma unroll
  for (int c = 0; c < 4; ++c) db2v[c] = wt[OFF_DB2 + 16*c + n];
  float db3 = wt[OFF_DB3];

  // zero-pad feats buffer so disc L1 A read can use uniform kvalid=32
  for (int t = lane; t < PP*32; t += 32) Fb[t] = 0.0f;

  // lane = cur peak j (read back what k_peaks wrote to d_out)
  float cxf = out[C_OFF + (b*PP + lane)*2 + 0];
  float cyf = out[C_OFF + (b*PP + lane)*2 + 1];
  float sj  = out[S_OFF + b*PP + lane];
  float mj  = out[M_OFF + b*PP + lane];
  float eold[8];
#pragma unroll
  for (int k = 0; k < 8; ++k) eold[k] = ws[WS_EMB_OLD + b*8 + k];

  // embedding B operands (invariant over i): single vector load each
  v16h bw1 = load_bop(hw, OPB_EW1, lane);
  v16h bw2 = load_bop(hw, OPB_EW2, lane);
  v16h bwf = load_bop(hw, OPB_EWF, lane);

  float* S = Ub;  // [32][16] embedding stage inside U
  wsync();
  for (int i = 0; i < PP; ++i) {
    float cxi = __shfl(cxf, i, 32), cyi = __shfl(cyf, i, 32), si = __shfl(sj, i, 32);
    // vss_cur row j = [c_i - c_j, s_i, s_j] kept in registers
    float f0 = cxi - cxf, f1 = cyi - cyf;
    // L1: 4 -> 8 (A from registers, no staging)
    v16h a0 = make_a_reg4(f0, f1, si, sj, 0, lane);
    v16h a1 = make_a_reg4(f0, f1, si, sj, 16, lane);
    v8f c0 = wmma16(a0, bw1, zero8()), c1 = wmma16(a1, bw1, zero8());
    wsync();
    store_c(S, 16, 0, c0, eb1, true, lane);
    store_c(S, 16, 16, c1, eb1, true, lane);
    wsync();
    a0 = make_a(S, 16, 0, 16, lane); a1 = make_a(S, 16, 16, 16, lane);
    c0 = wmma16(a0, bw2, zero8()); c1 = wmma16(a1, bw2, zero8());
    wsync();
    store_c(S, 16, 0, c0, eb2, true, lane);
    store_c(S, 16, 16, c1, eb2, true, lane);
    wsync();
    a0 = make_a(S, 16, 0, 16, lane); a1 = make_a(S, 16, 16, 16, lane);
    c0 = wmma16(a0, bwf, zero8()); c1 = wmma16(a1, bwf, zero8());
    wsync();
    store_c(S, 16, 0, c0, ebf, false, lane);
    store_c(S, 16, 16, c1, ebf, false, lane);
    wsync();

    // masked softmax over j
    float wl = S[lane*16 + 8];
    wl = (mj > 0.5f) ? wl : NEGL;
    float mx = wl;
    for (int off = 16; off; off >>= 1) mx = fmaxf(mx, __shfl_xor(mx, off, 32));
    float e = __expf(wl - mx), se = e;
    for (int off = 16; off; off >>= 1) se += __shfl_xor(se, off, 32);
    float wj = e / se;
    float e8[8];
#pragma unroll
    for (int k = 0; k < 8; ++k) e8[k] = wj * S[lane*16 + k];
    for (int off = 16; off; off >>= 1) {
#pragma unroll
      for (int k = 0; k < 8; ++k) e8[k] += __shfl_xor(e8[k], off, 32);
    }
    if (lane == 0) {
      Fb[i*32 + 0] = si;
#pragma unroll
      for (int k = 0; k < 8; ++k) Fb[i*32 + 1 + k] = e8[k];
#pragma unroll
      for (int k = 0; k < 8; ++k) Fb[i*32 + 9 + k] = eold[k];
    }
    wsync();
  }
  wsync();

  // ---- discriminator on 32 rows of feats17 ----
  // L1: 17 -> 32 (Fb cols 17..31 are zero -> uniform kvalid=32)
  v16h da0 = make_a(Fb, 32, 0, 32, lane), da1 = make_a(Fb, 32, 16, 32, lane);
  v16h db0 = load_bop(hw, OPB_DW1A, lane);
  v16h db1 = load_bop(hw, OPB_DW1B, lane);
  v8f c00 = wmma16(da0, db0, zero8()), c01 = wmma16(da0, db1, zero8());
  v8f c10 = wmma16(da1, db0, zero8()), c11 = wmma16(da1, db1, zero8());
  wsync();
  store_c(Fb,      32, 0,  c00, db1a, true, lane);
  store_c(Fb + 16, 32, 0,  c01, db1b, true, lane);
  store_c(Fb,      32, 16, c10, db1a, true, lane);
  store_c(Fb + 16, 32, 16, c11, db1b, true, lane);
  wsync();
  // L2: 32 -> 64 (A fully in registers: 2 operands cover all K)
  da0 = make_a(Fb, 32, 0, 32, lane); da1 = make_a(Fb, 32, 16, 32, lane);
  wsync();
#pragma unroll
  for (int nc = 0; nc < 4; ++nc) {
    v16h bb = load_bop(hw, OPB_DW2_0 + nc, lane);
    v8f h0 = wmma16(da0, bb, zero8());
    v8f h1 = wmma16(da1, bb, zero8());
    store_c(Ub + nc*16, 64, 0,  h0, db2v[nc], true, lane);
    store_c(Ub + nc*16, 64, 16, h1, db2v[nc], true, lane);
  }
  wsync();
  // L3: 64 -> 1, two chained K=32 WMMAs accumulating into C
  v16h w3a = load_bop(hw, OPB_DW3A, lane);
  v16h w3b = load_bop(hw, OPB_DW3B, lane);
  v16h a00 = make_a(Ub,      64, 0,  32, lane), a01 = make_a(Ub + 32, 64, 0,  32, lane);
  v16h a10 = make_a(Ub,      64, 16, 32, lane), a11 = make_a(Ub + 32, 64, 16, 32, lane);
  v8f m0 = wmma16(a01, w3b, wmma16(a00, w3a, zero8()));
  v8f m1 = wmma16(a11, w3b, wmma16(a10, w3a, zero8()));
  if ((lane & 15) == 0) {             // column N=0 holds the logits
    int rb = (lane & 16) ? 8 : 0;
#pragma unroll
    for (int r = 0; r < 8; ++r) {
      out[MP_OFF + b*PP + rb + r]      = m0[r] + db3;
      out[MP_OFF + b*PP + 16 + rb + r] = m1[r] + db3;
    }
  }
}

// ===========================================================================
extern "C" void kernel_launch(void* const* d_in, const int* in_sizes, int n_in,
                              void* d_out, int out_size, void* d_ws, size_t ws_size,
                              hipStream_t stream) {
  (void)in_sizes; (void)out_size; (void)ws_size;
  Params pr;
  for (int i = 0; i < NPAR; ++i)
    pr.p[i] = (const float*)d_in[(i < n_in) ? i : (n_in - 1)];
  const float* ts = (const float*)d_in[0];
  const float* an = (const float*)d_in[1];
  float* out = (float*)d_out;
  float* ws  = (float*)d_ws;

  k_prep <<<1, 256, 0, stream>>>(pr, ws);
  k_peaks<<<dim3(BT, 2), 256, 0, stream>>>(ts, out, ws);
  k_old  <<<BT / 8, 256, 0, stream>>>(an, ws);
  k_cur  <<<BT / 4, 128, 0, stream>>>(out, ws);
}
